// MultiheadSelfAttention_27865747816874
// MI455X (gfx1250) — matrix operand order repaired and checked by reading, not compiled, not run on hardware
//
#include <hip/hip_runtime.h>
#include <hip/hip_bf16.h>

typedef __bf16 bf16;
typedef __attribute__((ext_vector_type(16))) __bf16 bf16x16;
typedef __attribute__((ext_vector_type(8)))  __bf16 bf16x8;
typedef __attribute__((ext_vector_type(4)))  __bf16 bf16x4;
typedef __attribute__((ext_vector_type(8)))  float  f32x8;

#define DIM   1024
#define SEQ   2048
#define BATCH 4
#define HEADS 16
#define DK    64
#define MROWS (BATCH*SEQ)          // 8192
#define RTHETA 10000.0f

// ---- workspace layout (bytes) ----
#define XBF_OFF  ((size_t)0)                    // x  bf16: 8192*1024*2 = 16 MB
#define WQB_OFF  (XBF_OFF + (size_t)16777216)   // each w bf16: 2 MB
#define WKB_OFF  (WQB_OFF + (size_t)2097152)
#define WVB_OFF  (WKB_OFF + (size_t)2097152)
#define WOB_OFF  (WVB_OFF + (size_t)2097152)
#define QB_OFF   (WOB_OFF + (size_t)2097152)    // Q bf16 [B,S,D]   16 MB
#define KB_OFF   (QB_OFF  + (size_t)16777216)   // K bf16 [B,S,D]   16 MB
#define VT_OFF   (KB_OFF  + (size_t)16777216)   // V bf16 [B,H,DK,S] 16 MB (transposed)
#define OB_OFF   (VT_OFF  + (size_t)16777216)   // attn out bf16 [B,S,D] 16 MB

__device__ __forceinline__ f32x8 wmma_bf16(bf16x16 a, bf16x16 b, f32x8 c) {
  return __builtin_amdgcn_wmma_f32_16x16x32_bf16(false, a, false, b, (short)0, c,
                                                 false, false);
}

union FragU { bf16x16 v; bf16x8 h2[2]; };

// async copy of one 16-byte chunk: global -> LDS (ASYNCcnt path)
__device__ __forceinline__ void async_cp16(const bf16* gp, bf16* lp) {
  unsigned           lo = (unsigned)(uintptr_t)lp;                 // LDS byte address
  unsigned long long ga = (unsigned long long)(uintptr_t)gp;
  asm volatile("global_load_async_to_lds_b128 %0, %1, off"
               :: "v"(lo), "v"(ga) : "memory");
}

// ---------------- cast fp32 -> bf16, 4 elems/thread ----------------
__global__ void cast_f32_to_bf16(const float* __restrict__ in,
                                 bf16* __restrict__ out, int n4) {
  int i = blockIdx.x * blockDim.x + threadIdx.x;
  if (i >= n4) return;
  float4 v = ((const float4*)in)[i];
  bf16x4 o;
  o[0] = (bf16)v.x; o[1] = (bf16)v.y; o[2] = (bf16)v.z; o[3] = (bf16)v.w;
  ((bf16x4*)out)[i] = o;
}

// ---------------- WMMA GEMM:  out[m,e] = sum_d A[m,d] * W[e,d] ----------------
// MODE 0: Q proj (RoPE + 1/sqrt(dk) fold, bf16 out)
// MODE 1: K proj (RoPE, bf16 out)
// MODE 2: V proj (bf16 out, stored transposed as Vt[b][h][d][s])
// MODE 3: out proj (fp32 out)
template <int MODE>
__global__ __launch_bounds__(256) void gemm_wmma(const bf16* __restrict__ A,
                                                 const bf16* __restrict__ W,
                                                 bf16* __restrict__ outb,
                                                 float* __restrict__ outf) {
  const int lane = threadIdx.x & 31;
  const int wid  = threadIdx.x >> 5;     // 8 waves
  const int half = lane >> 4;
  const int l16  = lane & 15;
  const int m0 = blockIdx.y * 64  + (wid >> 2) * 32;   // wave tile 32 x 64
  const int n0 = blockIdx.x * 256 + (wid & 3) * 64;

  f32x8 acc[2][4] = {};

  for (int k = 0; k < DIM; k += 32) {
    FragU a[2];
#pragma unroll
    for (int mi = 0; mi < 2; mi++) {
      const bf16* ap = A + (size_t)(m0 + mi*16 + l16) * DIM + k + half * 8;
      a[mi].h2[0] = *(const bf16x8*)ap;          // K = half*8 + 0..7
      a[mi].h2[1] = *(const bf16x8*)(ap + 16);   // K = 16 + half*8 + 0..7
    }
#pragma unroll
    for (int ni = 0; ni < 4; ni++) {
      // B[k][n] = W[n_global][k_global]; per lane: row n0+ni*16+l16, K = half*8 + 0..7 and 16 + half*8 + 0..7
      FragU bu;
      const bf16* bp = W + (size_t)(n0 + ni*16 + l16) * DIM + k + half*8;
      bu.h2[0] = *(const bf16x8*)bp;
      bu.h2[1] = *(const bf16x8*)(bp + 16);
      bf16x16 b = bu.v;
#pragma unroll
      for (int mi = 0; mi < 2; mi++)
        acc[mi][ni] = wmma_bf16(a[mi].v, b, acc[mi][ni]);
    }
  }

#pragma unroll
  for (int mi = 0; mi < 2; mi++) {
#pragma unroll
    for (int ni = 0; ni < 4; ni++) {
      const int e = n0 + ni * 16 + l16;            // output feature (C layout: col = lane)
      if (MODE == 0 || MODE == 1) {
        const int dloc = e & (DK - 1);
        const float invf = __powf(RTHETA, -(float)(dloc & ~1) / (float)DK);
        const float qs = (MODE == 0) ? 0.125f : 1.0f;   // fold 1/sqrt(64) into Q
#pragma unroll
        for (int r = 0; r < 8; r++) {
          const int row  = m0 + mi * 16 + r + 8 * half; // C layout: row = reg + 8*half
          const int spos = row & (SEQ - 1);
          float sn, cs;
          __sincosf((float)spos * invf, &sn, &cs);
          const float x  = acc[mi][ni][r];
          const float pr = __shfl_xor(x, 1, 32);        // RoPE partner = adjacent lane
          const float o  = (e & 1) ? (pr * sn + x * cs) : (x * cs - pr * sn);
          outb[(size_t)row * DIM + e] = (bf16)(o * qs);
        }
      } else if (MODE == 2) {
        const int hh = e >> 6, d = e & 63;
#pragma unroll
        for (int r = 0; r < 8; r++) {
          const int row = m0 + mi * 16 + r + 8 * half;
          const int bb = row >> 11, s = row & (SEQ - 1);
          outb[(((size_t)bb * HEADS + hh) * DK + d) * SEQ + s] = (bf16)acc[mi][ni][r];
        }
      } else {
#pragma unroll
        for (int r = 0; r < 8; r++) {
          const int row = m0 + mi * 16 + r + 8 * half;
          outf[(size_t)row * DIM + e] = acc[mi][ni][r];
        }
      }
    }
  }
}

// ---------------- flash attention ----------------
// Block = 4 waves = 64 query rows of one (b,h). K/V 64-key tiles are staged once
// per block into LDS via GLOBAL_LOAD_ASYNC_TO_LDS_B128 and shared by all waves.
#define PSTRIDE 72   // P tile row stride (bf16), 144B
#define KSTRIDE 72   // K/V tile row stride (bf16), 144B: 16B-aligned, skews banks

__global__ __launch_bounds__(128) void attention_wmma(const bf16* __restrict__ Q,
                                                      const bf16* __restrict__ K,
                                                      const bf16* __restrict__ Vt,
                                                      bf16* __restrict__ O) {
  __shared__ __align__(16) bf16 sK[64 * KSTRIDE];          // [key][d]
  __shared__ __align__(16) bf16 sV[64 * KSTRIDE];          // [d][key]  (from Vt)
  __shared__ __align__(16) bf16 sP[4 * 16 * PSTRIDE];      // per-wave P tiles
  const int tid  = threadIdx.x;
  const int lane = tid & 31;
  const int wid  = tid >> 5;
  const int half = lane >> 4;
  const int l16  = lane & 15;
  const int b = blockIdx.z, h = blockIdx.y;
  const int qb0 = blockIdx.x * 64;
  const int q0  = qb0 + wid * 16;
  bf16* P = sP + wid * 16 * PSTRIDE;

  // Q fragments (A layout), resident for the whole loop; Q already scaled by 1/8
  FragU qa[2];
  {
    const size_t base = ((size_t)b * SEQ + q0 + l16) * DIM + h * DK;
#pragma unroll
    for (int kc = 0; kc < 2; kc++) {
      const bf16* qp = Q + base + kc * 32 + half * 8;
      qa[kc].h2[0] = *(const bf16x8*)qp;
      qa[kc].h2[1] = *(const bf16x8*)(qp + 16);
    }
  }

  f32x8 co[4] = {};
  float m[8], l[8];
#pragma unroll
  for (int r = 0; r < 8; r++) { m[r] = -1e30f; l[r] = 0.0f; }

  const size_t kgbase = (size_t)b * SEQ * DIM + h * DK;            // + key*DIM
  const size_t vgbase = ((size_t)b * HEADS + h) * DK * (size_t)SEQ; // + d*SEQ + key

  // Uniform trip count across the block (barriers inside): waves whose causal
  // window ends earlier see fully-masked tail blocks (exact no-op: exp->0).
  const int nkb = blockIdx.x + 1;
  for (int kb = 0; kb < nkb; kb++) {
    const int kbase = kb * 64;

    // ---- stage K tile [64 keys x 64 d] and V tile [64 d x 64 keys] into LDS ----
#pragma unroll
    for (int i = 0; i < 4; i++) {
      const int idx = tid + i * 128;          // 0..511: row = idx/8, 16B chunk = idx%8
      const int row = idx >> 3, c = idx & 7;
      async_cp16(K  + kgbase + (size_t)(kbase + row) * DIM + c * 8,
                 sK + row * KSTRIDE + c * 8);
      async_cp16(Vt + vgbase + (size_t)row * SEQ + kbase + c * 8,
                 sV + row * KSTRIDE + c * 8);
    }
    asm volatile("s_wait_asynccnt 0" ::: "memory");
    __syncthreads();

    // ---- scores: S(16q x 64k) = Q(16x64) . K^T, 4 N-tiles x 2 K-chunks ----
    f32x8 sc[4];
#pragma unroll
    for (int nt = 0; nt < 4; nt++) {
      const bf16* kp = sK + (nt * 16 + l16) * KSTRIDE + half * 8;
      FragU b0, b1;
      b0.h2[0] = *(const bf16x8*)kp;         b0.h2[1] = *(const bf16x8*)(kp + 16);
      b1.h2[0] = *(const bf16x8*)(kp + 32);  b1.h2[1] = *(const bf16x8*)(kp + 48);
      f32x8 c = {};
      c = wmma_bf16(qa[0].v, b0.v, c);
      c = wmma_bf16(qa[1].v, b1.v, c);
      sc[nt] = c;
    }

    if (kbase + 63 > q0) {   // causal mask (also handles fully-masked tail blocks)
#pragma unroll
      for (int nt = 0; nt < 4; nt++) {
        const int key = kbase + nt * 16 + l16;
#pragma unroll
        for (int r = 0; r < 8; r++)
          if (key > q0 + r + 8 * half) sc[nt][r] = -1000000.0f;  // reference NEG_INF
      }
    }

    // ---- online softmax stats (row lives across the 16 lanes of a half) ----
    float sf[8];
#pragma unroll
    for (int r = 0; r < 8; r++) {
      float t = fmaxf(fmaxf(sc[0][r], sc[1][r]), fmaxf(sc[2][r], sc[3][r]));
#pragma unroll
      for (int off = 1; off < 16; off <<= 1) t = fmaxf(t, __shfl_xor(t, off, 32));
      const float mn = fmaxf(m[r], t);
      sf[r] = __expf(m[r] - mn);
      m[r] = mn;
      l[r] *= sf[r];
    }
#pragma unroll
    for (int dt = 0; dt < 4; dt++)
#pragma unroll
      for (int r = 0; r < 8; r++) co[dt][r] *= sf[r];

    // ---- P = exp(S - m), row sums, repack P into wave-private LDS tile ----
#pragma unroll
    for (int nt = 0; nt < 4; nt++)
#pragma unroll
      for (int r = 0; r < 8; r++) {
        const float p = __expf(sc[nt][r] - m[r]);
        sc[nt][r] = p;
        P[(r + 8 * half) * PSTRIDE + nt * 16 + l16] = (bf16)p;
      }
#pragma unroll
    for (int r = 0; r < 8; r++) {
      float t = sc[0][r] + sc[1][r] + sc[2][r] + sc[3][r];
#pragma unroll
      for (int off = 1; off < 16; off <<= 1) t += __shfl_xor(t, off, 32);
      l[r] += t;
    }
    asm volatile("s_wait_dscnt 0" ::: "memory");   // wave-private LDS store->load

    // ---- A fragments of P from LDS ----
    FragU pa[2];
#pragma unroll
    for (int kc = 0; kc < 2; kc++) {
      const bf16* pp = P + l16 * PSTRIDE + kc * 32 + half * 8;
      pa[kc].h2[0] = *(const bf16x8*)pp;
      pa[kc].h2[1] = *(const bf16x8*)(pp + 16);
    }

    // ---- O(16x64) += P(16x64) . V(64x64), V B-frags contiguous in sV ----
#pragma unroll
    for (int dt = 0; dt < 4; dt++) {
      const bf16* vp = sV + (dt * 16 + l16) * KSTRIDE + half * 8;
      FragU v0, v1;
      v0.h2[0] = *(const bf16x8*)vp;         v0.h2[1] = *(const bf16x8*)(vp + 16);
      v1.h2[0] = *(const bf16x8*)(vp + 32);  v1.h2[1] = *(const bf16x8*)(vp + 48);
      co[dt] = wmma_bf16(pa[0].v, v0.v, co[dt]);
      co[dt] = wmma_bf16(pa[1].v, v1.v, co[dt]);
    }
    __syncthreads();   // all waves done with sK/sV before next stage overwrites
  }

  // ---- normalize and store attention output bf16 [B,S,D] ----
#pragma unroll
  for (int r = 0; r < 8; r++) {
    const float inv = 1.0f / l[r];
    const size_t obase = ((size_t)b * SEQ + q0 + r + 8 * half) * DIM + h * DK;
#pragma unroll
    for (int dt = 0; dt < 4; dt++)
      O[obase + dt * 16 + l16] = (bf16)(co[dt][r] * inv);
  }
}

// ---------------- host launch ----------------
extern "C" void kernel_launch(void* const* d_in, const int* in_sizes, int n_in,
                              void* d_out, int out_size, void* d_ws, size_t ws_size,
                              hipStream_t stream) {
  const float* x  = (const float*)d_in[0];
  const float* wq = (const float*)d_in[1];
  const float* wk = (const float*)d_in[2];
  const float* wv = (const float*)d_in[3];
  const float* wo = (const float*)d_in[4];
  float* out = (float*)d_out;

  char* ws = (char*)d_ws;
  bf16* xbf = (bf16*)(ws + XBF_OFF);
  bf16* wqb = (bf16*)(ws + WQB_OFF);
  bf16* wkb = (bf16*)(ws + WKB_OFF);
  bf16* wvb = (bf16*)(ws + WVB_OFF);
  bf16* wob = (bf16*)(ws + WOB_OFF);
  bf16* qb  = (bf16*)(ws + QB_OFF);
  bf16* kbp = (bf16*)(ws + KB_OFF);
  bf16* vt  = (bf16*)(ws + VT_OFF);
  bf16* ob  = (bf16*)(ws + OB_OFF);

  // casts
  {
    int n4 = (MROWS * DIM) / 4;   // 2097152
    cast_f32_to_bf16<<<(n4 + 255) / 256, 256, 0, stream>>>(x, xbf, n4);
    int w4 = (DIM * DIM) / 4;     // 262144
    cast_f32_to_bf16<<<(w4 + 255) / 256, 256, 0, stream>>>(wq, wqb, w4);
    cast_f32_to_bf16<<<(w4 + 255) / 256, 256, 0, stream>>>(wk, wkb, w4);
    cast_f32_to_bf16<<<(w4 + 255) / 256, 256, 0, stream>>>(wv, wvb, w4);
    cast_f32_to_bf16<<<(w4 + 255) / 256, 256, 0, stream>>>(wo, wob, w4);
  }

  dim3 ggrid(DIM / 256, MROWS / 64);   // (4, 128)
  gemm_wmma<0><<<ggrid, 256, 0, stream>>>(xbf, wqb, qb, nullptr);
  gemm_wmma<1><<<ggrid, 256, 0, stream>>>(xbf, wkb, kbp, nullptr);
  gemm_wmma<2><<<ggrid, 256, 0, stream>>>(xbf, wvb, vt, nullptr);

  attention_wmma<<<dim3(SEQ / 64, HEADS, BATCH), 128, 0, stream>>>(qb, kbp, vt, ob);

  gemm_wmma<3><<<ggrid, 256, 0, stream>>>(ob, wob, nullptr, out);
}
